// MultiHeadTemporalAttention_62328565400235
// MI455X (gfx1250) — matrix operand order repaired and checked by reading, not compiled, not run on hardware
//
#include <hip/hip_runtime.h>
#include <hip/hip_bf16.h>

// ---------------------------------------------------------------------------
// MultiHeadTemporalAttention fused flash-attention kernel for MI455X (gfx1250)
//
// Roofline reasoning (compile-only; reasoned from specs):
//   - K/V projection dominates: 2 x [131072,512]x[512,512] = 137 GFLOP.
//   - Fused (no K/V materialization): HBM traffic ~268 MB -> ~11.5 us @ 23.3TB/s.
//   - bf16 WMMA (16x16x32, f32 accum) -> ~27 us compute; f32 WMMA (K=4) would
//     be ~8x slower, scalar VALU ~100x. So: convert enc/W to bf16 in-flight
//     (native V_CVT_PK_BF16_F32), accumulate f32.
//   - B-fragment LDS layout interleaves K/V and n-tiles per lane-row so all
//     16 ds_load_b128 of a K-chunk use 16-bit immediate DS offsets from one
//     base register (no per-iteration address VALU).
// ---------------------------------------------------------------------------

#define B_   32
#define T_   4096
#define D_   512
#define H_   8
#define HD_  64
#define SCALE_ 0.125f  // 1/sqrt(64)

typedef __attribute__((ext_vector_type(16))) __bf16 v16bf;
typedef __attribute__((ext_vector_type(8)))  float  v8f;

union Frag16 {
  v16bf v;
  unsigned short u[16];
  uint4 q[2];
};

// Hardware bf16 convert (lowers to V_CVT_PK_BF16_F32 pairs).
__device__ __forceinline__ unsigned short f32_to_bf16_hw(float f) {
  __bf16 h = (__bf16)f;
  return __builtin_bit_cast(unsigned short, h);
}

__device__ __forceinline__ v8f wmma_bf16(v16bf a, v16bf b, v8f c) {
  // v_wmma_f32_16x16x32_bf16  (8-arg: neg_a, A, neg_b, B, c_mod, C, reuse_a, reuse_b)
  return __builtin_amdgcn_wmma_f32_16x16x32_bf16(false, a, false, b, (short)0, c,
                                                 false, false);
}

// ---------------------------------------------------------------------------
// Kernel 1: per-(b,h) fused projection + flash attention.
//   grid = B*H blocks of 128 threads (4 wave32s). Each wave owns 16 t-rows
//   per iteration (block covers 64 rows/iter, 64 iters over T=4096).
// ---------------------------------------------------------------------------
__global__ __launch_bounds__(128)
void attn_fused_kernel(const float* __restrict__ enc,
                       const float* __restrict__ Wq,
                       const float* __restrict__ Wk,
                       const float* __restrict__ Wv,
                       float* __restrict__ ws_scores,  // [B*H][T]
                       float* __restrict__ ws_m,       // [B*H]
                       float* __restrict__ ws_l,       // [B*H]
                       float* __restrict__ ws_ctx)     // [B*H][HD]
{
  // Interleaved B-matrix staging: kv_lds[lm(16)][nt(4)][mat(2: K,V)][Kdim(512)]
  // bf16. One lane base (lm*8192 B) reaches all of a K-chunk's 16 fragment
  // loads with immediate DS offsets (< 8.2 KB). 128 KB.
  __shared__ unsigned short kv_lds[16 * 4 * 2 * D_];
  __shared__ float q_lds[HD_];
  __shared__ float dump[4][16 * HD_];   // 16 KB (per-wave 16x64 f32 tile)
  __shared__ float mW[4], lW[4], ctxW[4][HD_];

  const int bh   = blockIdx.x;
  const int b    = bh / H_;
  const int h    = bh % H_;
  const int tid  = threadIdx.x;
  const int lane = tid & 31;
  const int wave = tid >> 5;
  const int hl   = lane >> 4;   // half of the wave (0/1)
  const int lm   = lane & 15;

  // ---- Stage Wk_h / Wv_h (rows h*64..h*64+63) into interleaved LDS ---------
  for (int i = tid; i < HD_ * D_; i += 128) {
    const int n = i >> 9;        // output column within head: 0..63
    const int k = i & (D_ - 1);  // K index: 0..511
    const int lmr = n & 15;      // lane-row
    const int ntr = n >> 4;      // n-tile
    const size_t gr = (size_t)(h * HD_ + n) * D_ + k;
    const int base = ((lmr * 4 + ntr) * 2) * D_ + k;
    kv_lds[base]      = f32_to_bf16_hw(Wk[gr]);   // mat 0 = K
    kv_lds[base + D_] = f32_to_bf16_hw(Wv[gr]);   // mat 1 = V
  }

  // ---- q = enc[b, T-1, :] @ Wq_h^T  (64 values, f32 exact) -----------------
  {
    const float* encLast = enc + ((size_t)b * T_ + (T_ - 1)) * D_;
    for (int n = tid; n < HD_; n += 128) {
      const float* wr = Wq + (size_t)(h * HD_ + n) * D_;
      float acc = 0.f;
      for (int d = 0; d < D_; ++d) acc += encLast[d] * wr[d];
      q_lds[n] = acc;
    }
  }
  __syncthreads();

  // ---- Flash loop over time tiles -----------------------------------------
  float m_run = -__builtin_inff();
  float l_run = 0.f;
  float ctx0 = 0.f, ctx1 = 0.f;      // context[lane], context[lane+32]
  float* mydump = dump[wave];
  // Per-lane B base: row lm of all (nt, mat) slices.
  const unsigned short* bbase = kv_lds + (size_t)lm * (4 * 2 * D_);

  for (int t0 = wave * 16; t0 < T_; t0 += 64) {
    // This lane's A row: enc[b, t0+lm, :]  (M = lm in the 16x32 A fragment)
    const float* arow = enc + ((size_t)b * T_ + t0 + lm) * D_;
    // Prefetch next iteration's row (global_prefetch_b8).
    __builtin_prefetch(arow + (size_t)64 * D_, 0, 1);

    v8f acck0 = {}, acck1 = {}, acck2 = {}, acck3 = {};
    v8f accv0 = {}, accv1 = {}, accv2 = {}, accv3 = {};

    for (int kc = 0; kc < D_ / 32; ++kc) {
      // A fragment (contiguous K, matching B):
      //   u[0..15] -> K = kc*32 + hl*16 + j
      const float* ap = arow + kc * 32 + hl * 16;
      const float4 a0 = *(const float4*)(ap + 0);
      const float4 a1 = *(const float4*)(ap + 4);
      const float4 a2 = *(const float4*)(ap + 8);
      const float4 a3 = *(const float4*)(ap + 12);
      Frag16 fa;
      fa.u[0]  = f32_to_bf16_hw(a0.x); fa.u[1]  = f32_to_bf16_hw(a0.y);
      fa.u[2]  = f32_to_bf16_hw(a0.z); fa.u[3]  = f32_to_bf16_hw(a0.w);
      fa.u[4]  = f32_to_bf16_hw(a1.x); fa.u[5]  = f32_to_bf16_hw(a1.y);
      fa.u[6]  = f32_to_bf16_hw(a1.z); fa.u[7]  = f32_to_bf16_hw(a1.w);
      fa.u[8]  = f32_to_bf16_hw(a2.x); fa.u[9]  = f32_to_bf16_hw(a2.y);
      fa.u[10] = f32_to_bf16_hw(a2.z); fa.u[11] = f32_to_bf16_hw(a2.w);
      fa.u[12] = f32_to_bf16_hw(a3.x); fa.u[13] = f32_to_bf16_hw(a3.y);
      fa.u[14] = f32_to_bf16_hw(a3.z); fa.u[15] = f32_to_bf16_hw(a3.w);

      // B fragment (32x16): lane = N, K = kc*32 + hl*16 + j (16 contiguous).
      // Address = bbase + (nt*2+mat)*512 + kb -> constant-folded DS offsets.
      const int kb = kc * 32 + hl * 16;
      auto loadB = [&](int nt, int mat) -> v16bf {
        Frag16 fb;
        const uint4* p = (const uint4*)(bbase + (nt * 2 + mat) * D_ + kb);
        fb.q[0] = p[0];
        fb.q[1] = p[1];
        return fb.v;
      };

      acck0 = wmma_bf16(fa.v, loadB(0, 0), acck0);
      acck1 = wmma_bf16(fa.v, loadB(1, 0), acck1);
      acck2 = wmma_bf16(fa.v, loadB(2, 0), acck2);
      acck3 = wmma_bf16(fa.v, loadB(3, 0), acck3);
      accv0 = wmma_bf16(fa.v, loadB(0, 1), accv0);
      accv1 = wmma_bf16(fa.v, loadB(1, 1), accv1);
      accv2 = wmma_bf16(fa.v, loadB(2, 1), accv2);
      accv3 = wmma_bf16(fa.v, loadB(3, 1), accv3);
    }

    // ---- dump K tile to wave-private LDS: C/D layout VGPR r -> M=r+8*hl ----
#pragma unroll
    for (int r = 0; r < 8; ++r) {
      const int row = r + 8 * hl;
      mydump[row * HD_ +  0 + lm] = acck0[r];
      mydump[row * HD_ + 16 + lm] = acck1[r];
      mydump[row * HD_ + 32 + lm] = acck2[r];
      mydump[row * HD_ + 48 + lm] = acck3[r];
    }
    __syncthreads();  // uniform trip counts across all 4 waves

    // ---- scores: s[row] = SCALE * sum_n q[n]*k[row][n]; row = lm -----------
    float partial = 0.f;
#pragma unroll
    for (int j = 0; j < 32; ++j) {
      const int n = hl * 32 + j;
      partial += q_lds[n] * mydump[lm * HD_ + n];
    }
    const float s = (partial + __shfl_xor(partial, 16, 32)) * SCALE_;

    // spill raw scaled scores for the weights.mean output
    if (hl == 0) ws_scores[(size_t)bh * T_ + t0 + lm] = s;

    // ---- online softmax update (per wave) ----------------------------------
    float tmax = s;
    tmax = fmaxf(tmax, __shfl_xor(tmax, 1, 32));
    tmax = fmaxf(tmax, __shfl_xor(tmax, 2, 32));
    tmax = fmaxf(tmax, __shfl_xor(tmax, 4, 32));
    tmax = fmaxf(tmax, __shfl_xor(tmax, 8, 32));
    const float m_new = fmaxf(m_run, tmax);
    const float alpha = __expf(m_run - m_new);
    const float p     = __expf(s - m_new);
    float psum = p;
    psum += __shfl_xor(psum, 1, 32);
    psum += __shfl_xor(psum, 2, 32);
    psum += __shfl_xor(psum, 4, 32);
    psum += __shfl_xor(psum, 8, 32);
    l_run = l_run * alpha + psum;
    m_run = m_new;

    // ---- dump V tile (reuse buffer) and accumulate context -----------------
#pragma unroll
    for (int r = 0; r < 8; ++r) {
      const int row = r + 8 * hl;
      mydump[row * HD_ +  0 + lm] = accv0[r];
      mydump[row * HD_ + 16 + lm] = accv1[r];
      mydump[row * HD_ + 32 + lm] = accv2[r];
      mydump[row * HD_ + 48 + lm] = accv3[r];
    }
    __syncthreads();

    ctx0 *= alpha;
    ctx1 *= alpha;
#pragma unroll
    for (int row = 0; row < 16; ++row) {
      const float pr = __shfl(p, row, 32);  // lane 'row' holds p for that row
      ctx0 += pr * mydump[row * HD_ + lane];
      ctx1 += pr * mydump[row * HD_ + lane + 32];
    }
  }

  // ---- merge the 4 waves ----------------------------------------------------
  ctxW[wave][lane]      = ctx0;
  ctxW[wave][lane + 32] = ctx1;
  if (lane == 0) { mW[wave] = m_run; lW[wave] = l_run; }
  __syncthreads();

  if (tid < HD_) {
    const float m = fmaxf(fmaxf(mW[0], mW[1]), fmaxf(mW[2], mW[3]));
    float L = 0.f, c = 0.f;
#pragma unroll
    for (int w = 0; w < 4; ++w) {
      const float e = __expf(mW[w] - m);
      L += lW[w] * e;
      c += ctxW[w][tid] * e;
    }
    ws_ctx[(size_t)bh * HD_ + tid] = c / L;
    if (tid == 0) { ws_m[bh] = m; ws_l[bh] = L; }
  }
}

// ---------------------------------------------------------------------------
// Kernel 2: output projection  out[b, :] = ctx[b, :] @ Wo.T    (tiny GEMM)
// ---------------------------------------------------------------------------
__global__ void out_proj_kernel(const float* __restrict__ ws_ctx,
                                const float* __restrict__ Wo,
                                float* __restrict__ out) {
  const int idx = blockIdx.x * blockDim.x + threadIdx.x;  // b*D + j
  if (idx >= B_ * D_) return;
  const int b = idx / D_, j = idx % D_;
  const float* c = ws_ctx + (size_t)b * D_;  // [B][H][HD] flattens to [B][D]
  const float* w = Wo + (size_t)j * D_;
  float acc = 0.f;
  for (int d = 0; d < D_; ++d) acc += c[d] * w[d];
  out[idx] = acc;
}

// ---------------------------------------------------------------------------
// Kernel 3: weights.mean(axis=heads) from spilled scores + softmax stats
// ---------------------------------------------------------------------------
__global__ void weights_mean_kernel(const float* __restrict__ ws_scores,
                                    const float* __restrict__ ws_m,
                                    const float* __restrict__ ws_l,
                                    float* __restrict__ out_w) {
  const int idx = blockIdx.x * blockDim.x + threadIdx.x;  // b*T + t
  if (idx >= B_ * T_) return;
  const int b = idx / T_, t = idx % T_;
  float acc = 0.f;
#pragma unroll
  for (int h = 0; h < H_; ++h) {
    const int bh = b * H_ + h;
    acc += __expf(ws_scores[(size_t)bh * T_ + t] - ws_m[bh]) / ws_l[bh];
  }
  out_w[idx] = acc * (1.f / H_);
}

// ---------------------------------------------------------------------------
extern "C" void kernel_launch(void* const* d_in, const int* in_sizes, int n_in,
                              void* d_out, int out_size, void* d_ws,
                              size_t ws_size, hipStream_t stream) {
  const float* enc = (const float*)d_in[0];
  const float* Wq  = (const float*)d_in[1];
  const float* Wk  = (const float*)d_in[2];
  const float* Wv  = (const float*)d_in[3];
  const float* Wo  = (const float*)d_in[4];
  float* out = (float*)d_out;

  // Workspace layout (f32): scores[B*H*T] | m[B*H] | l[B*H] | ctx[B*H*HD]
  float* ws        = (float*)d_ws;
  float* ws_scores = ws;                                   // 1,048,576
  float* ws_m      = ws_scores + (size_t)B_ * H_ * T_;     // 256
  float* ws_l      = ws_m + B_ * H_;                       // 256
  float* ws_ctx    = ws_l + B_ * H_;                       // 16,384

  attn_fused_kernel<<<B_ * H_, 128, 0, stream>>>(enc, Wq, Wk, Wv, ws_scores,
                                                 ws_m, ws_l, ws_ctx);
  out_proj_kernel<<<(B_ * D_ + 255) / 256, 256, 0, stream>>>(ws_ctx, Wo, out);
  weights_mean_kernel<<<(B_ * T_ + 255) / 256, 256, 0, stream>>>(
      ws_scores, ws_m, ws_l, out + B_ * D_);
}
